// CorrBlock_34608846471921
// MI455X (gfx1250) — matrix operand order repaired and checked by reading, not compiled, not run on hardware
//
#include <hip/hip_runtime.h>

// ---------------------------------------------------------------------------
// CorrBlock for RAFT-style optical flow on gfx1250 (MI455X).
//   B=4, D=256, H=W=64, num_levels=4, radius=4
// Pipeline:
//   1) transpose+convert fmap1/fmap2: [B,D,HW] f32 -> [B,HW,D] bf16 (LDS tile)
//   2) corr GEMM via v_wmma_f32_16x16x32_bf16 (wave tile 32x64, block 128x128)
//   3) 2x2 avg-pool pyramid (3 levels)
//   4) 81-point bilinear lookup per query per level
// ---------------------------------------------------------------------------

typedef __attribute__((ext_vector_type(16))) __bf16 v16bf;
typedef __attribute__((ext_vector_type(8)))  __bf16 v8bf;
typedef __attribute__((ext_vector_type(8)))  float  v8f;

#define BATCH 4
#define DK    256
#define HH    64
#define WW    64
#define HW    4096        // 64*64
#define NQ    16384       // BATCH*HW
#define NLVL  4
#define RAD   4
#define WIN   9           // 2*RAD+1
#define NCH   (NLVL*WIN*WIN)  // 324

// float -> bf16 (round-to-nearest-even)
__device__ __forceinline__ unsigned short f2bf(float f) {
    unsigned int u = __float_as_uint(f);
    u += 0x7FFFu + ((u >> 16) & 1u);
    return (unsigned short)(u >> 16);
}

// -------------------------------------------------------------------
// 1) Transpose+convert: src [B, D, HW] f32 -> dst [B, HW, D] bf16
// -------------------------------------------------------------------
__global__ __launch_bounds__(1024)
void k_transpose_bf16(const float* __restrict__ src, unsigned short* __restrict__ dst)
{
    __shared__ float tile[32][33];
    const int tx = threadIdx.x, ty = threadIdx.y;
    const int n0 = blockIdx.x * 32;   // HW tile base
    const int d0 = blockIdx.y * 32;   // D tile base
    const int b  = blockIdx.z;

    // coalesced read along HW
    tile[ty][tx] = src[((size_t)b * DK + (d0 + ty)) * HW + (n0 + tx)];
    __syncthreads();
    // coalesced write along D
    dst[((size_t)b * HW + (n0 + ty)) * DK + (d0 + tx)] = f2bf(tile[tx][ty]);
}

// -------------------------------------------------------------------
// 2) corr[b, m, n] = (1/16) * sum_d f1t[b,m,d] * f2t[b,n,d]
//    Block = 8 waves arranged 4(M) x 2(N); block tile 128(M) x 128(N).
//    Wave tile 32(M) x 64(N): 2 A-frags x 4 B-frags -> 8 WMMAs per k-step.
// -------------------------------------------------------------------
__global__ __launch_bounds__(256)
void k_corr_gemm(const unsigned short* __restrict__ f1t,
                 const unsigned short* __restrict__ f2t,
                 float* __restrict__ corr)
{
    const int lane = threadIdx.x & 31;
    const int wave = threadIdx.x >> 5;
    const int half = lane >> 4;       // lane half selects K sub-range
    const int l15  = lane & 15;
    const int mw   = wave & 3;        // wave position in block: 4 x 2
    const int nw   = wave >> 2;
    const int b    = blockIdx.z;
    const int m0   = blockIdx.x * 128 + mw * 32;
    const int n0   = blockIdx.y * 128 + nw * 64;

    // A rows for this lane: M = m0 + l15 and m0 + 16 + l15, contiguous over K
    const __bf16* A0 = (const __bf16*)f1t + ((size_t)b * HW + (m0 + l15)) * DK;
    const __bf16* A1 = A0 + (size_t)16 * DK;
    const __bf16* Bb = (const __bf16*)f2t + ((size_t)b * HW) * DK;

    v8f acc[2][4] = {{v8f{}, v8f{}, v8f{}, v8f{}},
                     {v8f{}, v8f{}, v8f{}, v8f{}}};

    for (int k0 = 0; k0 < DK; k0 += 32) {
        // ---- issue ALL loads for this k-step first (batched clauses) ----
        // A fragments (contiguous K, matching B):
        //   elems 0..7  -> K = k0 + 16*half + [0..7]
        //   elems 8..15 -> K = k0 + 16*half + 8 + [0..7]
        v8bf a0lo = *(const v8bf*)(A0 + k0 + 16 * half);
        v8bf a0hi = *(const v8bf*)(A0 + k0 + 16 * half + 8);
        v8bf a1lo = *(const v8bf*)(A1 + k0 + 16 * half);
        v8bf a1hi = *(const v8bf*)(A1 + k0 + 16 * half + 8);

        // B fragments (contiguous K, same order as A): lane holds N = n0+16t+l15,
        // 16 contiguous K starting at k0 + 16*half
        v8bf blo[4], bhi[4];
        #pragma unroll
        for (int t = 0; t < 4; ++t) {
            const __bf16* bp = Bb + ((size_t)(n0 + 16 * t + l15)) * DK + k0 + 16 * half;
            blo[t] = *(const v8bf*)(bp);
            bhi[t] = *(const v8bf*)(bp + 8);
        }
        if (k0 + 32 < DK) {
            __builtin_prefetch(A0 + k0 + 32, 0, 1);   // global_prefetch_b8
            __builtin_prefetch(A1 + k0 + 32, 0, 1);
        }

        v16bf af[2];
        af[0] = __builtin_shufflevector(a0lo, a0hi,
            0,1,2,3,4,5,6,7,8,9,10,11,12,13,14,15);
        af[1] = __builtin_shufflevector(a1lo, a1hi,
            0,1,2,3,4,5,6,7,8,9,10,11,12,13,14,15);

        // ---- 8 WMMAs consuming the batch ----
        #pragma unroll
        for (int t = 0; t < 4; ++t) {
            v16bf bf = __builtin_shufflevector(blo[t], bhi[t],
                0,1,2,3,4,5,6,7,8,9,10,11,12,13,14,15);
            #pragma unroll
            for (int mi = 0; mi < 2; ++mi) {
                acc[mi][t] = __builtin_amdgcn_wmma_f32_16x16x32_bf16(
                    false, af[mi], false, bf, (short)0, acc[mi][t], false, false);
            }
        }
    }

    // C/D layout: lanes 0-15 -> N=lane, M=r ; lanes 16-31 -> N=lane-16, M=r+8
    const float scale = 0.0625f;   // 1/sqrt(256)
    #pragma unroll
    for (int mi = 0; mi < 2; ++mi) {
        const size_t rowbase = (size_t)b * HW + m0 + 16 * mi + 8 * half;
        #pragma unroll
        for (int t = 0; t < 4; ++t) {
            const int col = n0 + 16 * t + l15;
            #pragma unroll
            for (int r = 0; r < 8; ++r) {
                corr[(rowbase + r) * (size_t)HW + col] = acc[mi][t][r] * scale;
            }
        }
    }
}

// -------------------------------------------------------------------
// 3) 2x2 average pool:  src [NQ, 2h, 2w] -> dst [NQ, h, w]
// -------------------------------------------------------------------
__global__ __launch_bounds__(256)
void k_pool2x2(const float* __restrict__ src, float* __restrict__ dst,
               int h, int w, long long total)
{
    long long tid = (long long)blockIdx.x * blockDim.x + threadIdx.x;
    if (tid >= total) return;
    const int hw = h * w;
    long long q = tid / hw;
    int rem = (int)(tid - q * hw);
    int yy = rem / w;
    int xx = rem - yy * w;
    const int w2 = 2 * w;
    const float* s = src + q * (size_t)(4 * hw) + (size_t)(2 * yy) * w2 + 2 * xx;
    dst[tid] = 0.25f * (s[0] + s[1] + s[w2] + s[w2 + 1]);
}

// -------------------------------------------------------------------
// 4) Bilinear lookup.  One thread per (query, channel).
//    channel c = lvl*81 + a*9 + bb ; sample at (x/2^lvl + (a-4), y/2^lvl + (bb-4))
// -------------------------------------------------------------------
__device__ __forceinline__ float corner_val(const float* img, int Wl, int Hl,
                                            int yi, int xi)
{
    if (xi < 0 || xi > Wl - 1 || yi < 0 || yi > Hl - 1) return 0.0f;
    return img[yi * Wl + xi];
}

__global__ __launch_bounds__(256)
void k_lookup(const float* __restrict__ c0, const float* __restrict__ c1,
              const float* __restrict__ c2, const float* __restrict__ c3,
              const float* __restrict__ coords, float* __restrict__ out)
{
    long long tid = (long long)blockIdx.x * blockDim.x + threadIdx.x;
    const long long total = (long long)NQ * NCH;
    if (tid >= total) return;

    const int c = (int)(tid % NCH);
    const int q = (int)(tid / NCH);          // q = b*4096 + y*64 + x
    const int b = q >> 12;
    const int p = q & (HW - 1);

    const int lvl = c / (WIN * WIN);
    const int wi  = c - lvl * (WIN * WIN);
    const int a   = wi / WIN;                // added to x
    const int bb  = wi - a * WIN;            // added to y

    const float cx = coords[((size_t)b * 2 + 0) * HW + p];
    const float cy = coords[((size_t)b * 2 + 1) * HW + p];

    const float inv = 1.0f / (float)(1 << lvl);
    const float xs = cx * inv + (float)(a - RAD);
    const float ys = cy * inv + (float)(bb - RAD);

    const float* img;
    int Wl;
    if      (lvl == 0) { img = c0; Wl = 64; }
    else if (lvl == 1) { img = c1; Wl = 32; }
    else if (lvl == 2) { img = c2; Wl = 16; }
    else               { img = c3; Wl = 8;  }
    const int Hl = Wl;
    img += (size_t)q * (size_t)(Wl * Wl);

    const float x0f = floorf(xs);
    const float y0f = floorf(ys);
    const int x0 = (int)x0f;
    const int y0 = (int)y0f;
    const float wx = xs - x0f;
    const float wy = ys - y0f;

    const float v00 = corner_val(img, Wl, Hl, y0,     x0);
    const float v01 = corner_val(img, Wl, Hl, y0,     x0 + 1);
    const float v10 = corner_val(img, Wl, Hl, y0 + 1, x0);
    const float v11 = corner_val(img, Wl, Hl, y0 + 1, x0 + 1);

    const float val = v00 * (1.0f - wy) * (1.0f - wx)
                    + v01 * (1.0f - wy) * wx
                    + v10 * wy * (1.0f - wx)
                    + v11 * wy * wx;

    out[((size_t)b * NCH + c) * HW + p] = val;
}

// ---------------------------------------------------------------------------
extern "C" void kernel_launch(void* const* d_in, const int* in_sizes, int n_in,
                              void* d_out, int out_size, void* d_ws, size_t ws_size,
                              hipStream_t stream)
{
    const float* fmap1  = (const float*)d_in[0];
    const float* fmap2  = (const float*)d_in[1];
    const float* coords = (const float*)d_in[2];
    // d_in[3]=num_levels(4), d_in[4]=radius(4): fixed by reference setup.

    // Workspace layout
    float* c0 = (float*)d_ws;                       // [NQ, 64, 64]
    float* c1 = c0 + (size_t)NQ * 4096;             // [NQ, 32, 32]
    float* c2 = c1 + (size_t)NQ * 1024;             // [NQ, 16, 16]
    float* c3 = c2 + (size_t)NQ * 256;              // [NQ,  8,  8]
    unsigned short* f1t = (unsigned short*)(c3 + (size_t)NQ * 64);
    unsigned short* f2t = f1t + (size_t)BATCH * HW * DK;

    // 1) transpose + bf16 convert both feature maps
    {
        dim3 grid(HW / 32, DK / 32, BATCH), block(32, 32, 1);
        k_transpose_bf16<<<grid, block, 0, stream>>>(fmap1, f1t);
        k_transpose_bf16<<<grid, block, 0, stream>>>(fmap2, f2t);
    }

    // 2) correlation volume GEMM (WMMA bf16 -> f32), block tile 128x128
    {
        dim3 grid(HW / 128, HW / 128, BATCH), block(256, 1, 1);
        k_corr_gemm<<<grid, block, 0, stream>>>(f1t, f2t, c0);
    }

    // 3) pyramid
    {
        long long t1 = (long long)NQ * 32 * 32;
        long long t2 = (long long)NQ * 16 * 16;
        long long t3 = (long long)NQ * 8 * 8;
        k_pool2x2<<<(int)((t1 + 255) / 256), 256, 0, stream>>>(c0, c1, 32, 32, t1);
        k_pool2x2<<<(int)((t2 + 255) / 256), 256, 0, stream>>>(c1, c2, 16, 16, t2);
        k_pool2x2<<<(int)((t3 + 255) / 256), 256, 0, stream>>>(c2, c3, 8, 8, t3);
    }

    // 4) bilinear lookup -> d_out [B, 324, H, W]
    {
        long long total = (long long)NQ * NCH;
        k_lookup<<<(int)((total + 255) / 256), 256, 0, stream>>>(
            c0, c1, c2, c3, coords, (float*)d_out);
    }
}
